// HGTLayer_87737591923455
// MI455X (gfx1250) — matrix operand order repaired and checked by reading, not compiled, not run on hardware
//
#include <hip/hip_runtime.h>
#include <hip/hip_bf16.h>

typedef _Float16 half_t;
typedef __attribute__((ext_vector_type(16))) _Float16 v16h;
typedef __attribute__((ext_vector_type(8)))  _Float16 v8h;
typedef __attribute__((ext_vector_type(8)))  float    v8f;
typedef __attribute__((ext_vector_type(4)))  unsigned int u32x4;
typedef __attribute__((ext_vector_type(8)))  int      i32x8;
typedef __attribute__((ext_vector_type(4)))  int      i32x4;

#define OUTD 128
#define RSQRT_DK 0.17677669529663687f  // 1/sqrt(32)
#define A_PITCH 136                    // halfs per LDS row: 128 + 8 (16B TDM pad)
#define HB_PITCH 132                   // floats per LN-staging row (conflict pad)

// ---------------------------------------------------------------------------
// f32 -> f16 convert
// ---------------------------------------------------------------------------
__global__ void cvt_f32_f16(const float* __restrict__ in, half_t* __restrict__ out, int n) {
    int i = blockIdx.x * blockDim.x + threadIdx.x;
    if (i < n) out[i] = (half_t)in[i];
}

// ---------------------------------------------------------------------------
// Pre-shuffle a 128x128 f32 weight (row-major W[k][n]) into B-fragment order
// for v_wmma_f32_16x16x32_f16:
//   out[((nstrip*4 + kstep)*32 + lane)*16 + e] = W[k][n]
//   n = nstrip*16 + lane%16 ;  k = kstep*32 + (lane<16 ? 0 : 16) + e
// ---------------------------------------------------------------------------
__global__ void prep_wfrag(const float* __restrict__ W, half_t* __restrict__ out) {
    int idx = blockIdx.x * blockDim.x + threadIdx.x;
    if (idx >= 128 * 128) return;
    int e    = idx & 15;
    int lane = (idx >> 4) & 31;
    int ks   = (idx >> 9) & 3;
    int ns   = (idx >> 11) & 7;
    int n = ns * 16 + (lane & 15);
    int k = ks * 32 + ((lane < 16) ? 0 : 16) + e;
    out[idx] = (half_t)W[k * 128 + n];
}

// ---------------------------------------------------------------------------
// TDM: stage the block's 64x128 f16 A-tile into LDS (offset 0) with a 16B pad
// after every 256B row (LDS pitch = 272B = A_PITCH halfs). Rows past M are
// zero-filled by TDM OOB handling (tensor_dim1 = M - m0).
// Issued by wave 0 only; TENSORcnt waited by wave 0, then workgroup barrier.
// ---------------------------------------------------------------------------
__device__ __forceinline__ void tdm_load_A(const half_t* __restrict__ A, int M, int m0) {
    unsigned long long ga = (unsigned long long)(const void*)(A + (size_t)m0 * OUTD);
    unsigned dim1 = (unsigned)(M - m0);          // remaining rows (OOB rows -> zero)

    u32x4 g0;
    g0[0] = 1u;                                   // count=1, load descriptor
    g0[1] = 0u;                                   // lds_addr (single smem block @0)
    g0[2] = (unsigned)(ga & 0xFFFFFFFFu);         // global_addr[31:0]
    g0[3] = (unsigned)((ga >> 32) & 0x1FFFFFFu)   // global_addr[56:32]
          | (2u << 30);                           // type = 2 ("image")

    // data_size=2B(1), pad_enable, pad_interval=64 DW (5), pad_amount=4 DW (3)
    unsigned d0 = (1u << 16) | (1u << 20) | (5u << 22) | (3u << 25);
    unsigned dim0 = 128u, tile0 = 128u, tile1 = 64u, stride0 = 128u;
    i32x8 g1;
    g1[0] = (int)d0;
    g1[1] = (int)((dim0 & 0xFFFFu) << 16);                       // tensor_dim0[15:0]
    g1[2] = (int)(((dim0 >> 16) & 0xFFFFu) | ((dim1 & 0xFFFFu) << 16));
    g1[3] = (int)(((dim1 >> 16) & 0xFFFFu) | (tile0 << 16));     // tile_dim0
    g1[4] = (int)tile1;                                          // tile_dim1 (tile_dim2=0)
    g1[5] = (int)stride0;                                        // tensor_dim0_stride[31:0]
    g1[6] = 0;
    g1[7] = 0;

    i32x4 gz4 = {0, 0, 0, 0};
    i32x8 gz8 = {0, 0, 0, 0, 0, 0, 0, 0};
    __builtin_amdgcn_tensor_load_to_lds(g0, g1, gz4, gz4, gz8, 0);
    __builtin_amdgcn_s_wait_tensorcnt(0);
}

// Make TDM-written LDS visible to the optimizer: the pointer escapes into an
// opaque asm that may write memory, so loads from it cannot fold to poison.
__device__ __forceinline__ void lds_written_by_tdm(void* p) {
    asm volatile("" : : "v"(p) : "memory");
}

// ---------------------------------------------------------------------------
// Core WMMA tile compute: 64 rows x 128 cols per 256-thread block.
// A fragments come from the TDM-staged LDS tile; B frags live in registers.
// ---------------------------------------------------------------------------
__device__ __forceinline__ void gemm_core_lds(const half_t* __restrict__ Albs,
                                              const half_t* __restrict__ Wfrag,
                                              int wave, int lane, v8f acc[4]) {
    v16h bfrag[4];
#pragma unroll
    for (int ks = 0; ks < 4; ++ks)
        bfrag[ks] = *(const v16h*)(Wfrag + (size_t)(((wave * 4 + ks) * 32) + lane) * 16);

#pragma unroll
    for (int mt = 0; mt < 4; ++mt)
#pragma unroll
        for (int i = 0; i < 8; ++i) acc[mt][i] = 0.f;

    const int halfSel   = (lane < 16) ? 0 : 16;
    const int rowInTile = lane & 15;

#pragma unroll
    for (int mt = 0; mt < 4; ++mt) {
        const half_t* arow = Albs + (mt * 16 + rowInTile) * A_PITCH;
#pragma unroll
        for (int ks = 0; ks < 4; ++ks) {
            int c0 = ks * 32 + halfSel;
            v8h lo = *(const v8h*)(arow + c0);
            v8h hi = *(const v8h*)(arow + c0 + 8);
            v16h af;
#pragma unroll
            for (int i = 0; i < 8; ++i) { af[i] = lo[i]; af[i + 8] = hi[i]; }
            acc[mt] = __builtin_amdgcn_wmma_f32_16x16x32_f16(
                false, af, false, bfrag[ks], (short)0, acc[mt], false, false);
        }
    }
}

// GEMM -> f16 output (+ optional bias). Used for K/Q/V projections and agg@Wm+bm.
__global__ __launch_bounds__(256)
void gemm128_wmma_h(const half_t* __restrict__ A, const half_t* __restrict__ Wfrag,
                    const float* __restrict__ bias, half_t* __restrict__ Ch, int M) {
    __shared__ half_t Abuf[64 * A_PITCH];        // 17408 B @ LDS offset 0
    const int tid  = threadIdx.x;
    const int wave = tid >> 5;
    const int lane = tid & 31;
    const int m0   = blockIdx.x * 64;

    if (tid < 32) tdm_load_A(A, M, m0);          // one TDM op per block
    lds_written_by_tdm(Abuf);
    __syncthreads();

    v8f acc[4];
    gemm_core_lds(Abuf, Wfrag, wave, lane, acc);

    const int col = wave * 16 + (lane & 15);
    const float b = bias ? bias[col] : 0.f;
#pragma unroll
    for (int mt = 0; mt < 4; ++mt) {
        int rbase = m0 + mt * 16 + ((lane < 16) ? 0 : 8);
#pragma unroll
        for (int r = 0; r < 8; ++r) {
            int row = rbase + r;
            if (row < M) Ch[(size_t)row * OUTD + col] = (half_t)(acc[mt][r] + b);
        }
    }
}

// GEMM + bias + residual + LayerNorm fused epilogue -> f32 output.
__global__ __launch_bounds__(256)
void gemm128_wmma_ln(const half_t* __restrict__ A, const half_t* __restrict__ Wfrag,
                     const float* __restrict__ bias, const float* __restrict__ xres,
                     const float* __restrict__ gamma, const float* __restrict__ beta,
                     float* __restrict__ out, int M) {
    __shared__ unsigned char smem[64 * A_PITCH * 2 + 64 * HB_PITCH * 4]; // 17408 + 33792 B
    half_t* Abuf = (half_t*)smem;                // must sit at LDS offset 0 (TDM target)
    float*  hb   = (float*)(smem + 64 * A_PITCH * 2);

    const int tid  = threadIdx.x;
    const int wave = tid >> 5;
    const int lane = tid & 31;
    const int m0   = blockIdx.x * 64;

    if (tid < 32) tdm_load_A(A, M, m0);
    lds_written_by_tdm(smem);
    __syncthreads();

    v8f acc[4];
    gemm_core_lds(Abuf, Wfrag, wave, lane, acc);

    const int col = wave * 16 + (lane & 15);
    const float b = bias[col];
#pragma unroll
    for (int mt = 0; mt < 4; ++mt) {
        int rl = mt * 16 + ((lane < 16) ? 0 : 8);
#pragma unroll
        for (int r = 0; r < 8; ++r) {
            int rr   = rl + r;
            int grow = m0 + rr;
            if (grow >= M) grow = M - 1;
            hb[rr * HB_PITCH + col] = acc[mt][r] + b + xres[(size_t)grow * OUTD + col];
        }
    }
    __syncthreads();

    // LayerNorm: 4 threads per row (each covers 32 cols), reduce via shfl_xor.
    const int r = tid >> 2;
    const int q = tid & 3;
    const float* hr = hb + r * HB_PITCH + q * 32;
    float sum = 0.f, ss = 0.f;
#pragma unroll
    for (int c = 0; c < 32; ++c) { float v = hr[c]; sum += v; ss += v * v; }
    sum += __shfl_xor(sum, 1, 32);  ss += __shfl_xor(ss, 1, 32);
    sum += __shfl_xor(sum, 2, 32);  ss += __shfl_xor(ss, 2, 32);
    const float mean = sum * (1.f / OUTD);
    const float var  = ss * (1.f / OUTD) - mean * mean;
    const float rstd = rsqrtf(var + 1e-5f);
    const int grow = m0 + r;
    if (grow < M) {
#pragma unroll
        for (int c = 0; c < 32; ++c) {
            int cc = q * 32 + c;
            out[(size_t)grow * OUTD + cc] =
                (hb[r * HB_PITCH + cc] - mean) * rstd * gamma[cc] + beta[cc];
        }
    }
}

// ---------------------------------------------------------------------------
// Per-node attention bias: a[node][h] = sum_k K[node][k] * Wa[k][h]   (H=4)
// ---------------------------------------------------------------------------
__global__ void attn_alpha(const half_t* __restrict__ Kh, const float* __restrict__ Wa,
                           float* __restrict__ A, int N) {
    int idx = blockIdx.x * blockDim.x + threadIdx.x;
    if (idx >= N * 4) return;
    int node = idx >> 2, h = idx & 3;
    const half_t* kp = Kh + (size_t)node * OUTD;
    float s = 0.f;
    for (int k = 0; k < OUTD; ++k) s += (float)kp[k] * Wa[k * 4 + h];
    A[idx] = s;
}

// ---------------------------------------------------------------------------
// Edge kernel: one wave per edge. Lane l owns dims [4l,4l+4) -> head l/8.
// Softmax is over the 4 heads, so wsum contribution per edge is exactly 1.
// ---------------------------------------------------------------------------
__global__ __launch_bounds__(256)
void edge_attn(const half_t* __restrict__ Ksrc, const half_t* __restrict__ Vsrc,
               const half_t* __restrict__ Qdst, const float* __restrict__ Asrc,
               const int* __restrict__ ei, int E,
               float* __restrict__ msg, float* __restrict__ wsum) {
    int wid  = (blockIdx.x * blockDim.x + threadIdx.x) >> 5;
    int lane = threadIdx.x & 31;
    if (wid >= E) return;                    // uniform per wave
    const int src = ei[wid];
    const int dst = ei[E + wid];

    const half_t* kp = Ksrc + (size_t)src * OUTD + lane * 4;
    const half_t* qp = Qdst + (size_t)dst * OUTD + lane * 4;
    const half_t* vp = Vsrc + (size_t)src * OUTD + lane * 4;

    float dot = 0.f, v[4];
#pragma unroll
    for (int i = 0; i < 4; ++i) {
        dot += (float)kp[i] * (float)qp[i];
        v[i] = (float)vp[i];
    }
    // reduce within 8-lane head groups
    dot += __shfl_xor(dot, 1, 32);
    dot += __shfl_xor(dot, 2, 32);
    dot += __shfl_xor(dot, 4, 32);

    const int h = lane >> 3;
    float s = dot * RSQRT_DK + Asrc[(size_t)src * 4 + h];

    float s0 = __shfl(s, 0, 32), s1 = __shfl(s, 8, 32);
    float s2 = __shfl(s, 16, 32), s3 = __shfl(s, 24, 32);
    float m  = fmaxf(fmaxf(s0, s1), fmaxf(s2, s3));
    float e0 = __expf(s0 - m), e1 = __expf(s1 - m);
    float e2 = __expf(s2 - m), e3 = __expf(s3 - m);
    float inv = 1.f / (e0 + e1 + e2 + e3);
    float my  = ((h == 0) ? e0 : (h == 1) ? e1 : (h == 2) ? e2 : e3) * inv;

    float* mp = msg + (size_t)dst * OUTD + lane * 4;
#pragma unroll
    for (int i = 0; i < 4; ++i) atomicAdd(mp + i, my * v[i]);
    if (lane == 0) atomicAdd(wsum + dst, 1.0f);
}

// ---------------------------------------------------------------------------
// agg = msg / max(wsum,1e-8), converted to f16 for the next GEMM.
// ---------------------------------------------------------------------------
__global__ void agg_norm(const float* __restrict__ msg, const float* __restrict__ wsum,
                         half_t* __restrict__ out, int N) {
    int idx = blockIdx.x * blockDim.x + threadIdx.x;
    if (idx >= N * OUTD) return;
    int node = idx >> 7;
    float w = fmaxf(wsum[node], 1e-8f);
    out[idx] = (half_t)(msg[idx] / w);
}

// ---------------------------------------------------------------------------
extern "C" void kernel_launch(void* const* d_in, const int* in_sizes, int n_in,
                              void* d_out, int out_size, void* d_ws, size_t ws_size,
                              hipStream_t stream) {
    const float* x_user  = (const float*)d_in[0];
    const float* x_item  = (const float*)d_in[1];
    const int*   ei_ui   = (const int*)d_in[2];
    const int*   ei_iu   = (const int*)d_in[3];
    const float* Wk_u    = (const float*)d_in[4];
    const float* Wq_u    = (const float*)d_in[5];
    const float* Wv_u    = (const float*)d_in[6];
    const float* Wk_i    = (const float*)d_in[7];
    const float* Wq_i    = (const float*)d_in[8];
    const float* Wv_i    = (const float*)d_in[9];
    const float* Wa_ui   = (const float*)d_in[10];
    const float* Wa_iu   = (const float*)d_in[11];
    const float* Wm_u    = (const float*)d_in[12];
    const float* bm_u    = (const float*)d_in[13];
    const float* Wm_i    = (const float*)d_in[14];
    const float* bm_i    = (const float*)d_in[15];
    const float* Wg_u    = (const float*)d_in[16];
    const float* bg_u    = (const float*)d_in[17];
    const float* Wg_i    = (const float*)d_in[18];
    const float* bg_i    = (const float*)d_in[19];
    const float* lng_u   = (const float*)d_in[20];
    const float* lnb_u   = (const float*)d_in[21];
    const float* lng_i   = (const float*)d_in[22];
    const float* lnb_i   = (const float*)d_in[23];

    const int Nu = in_sizes[0] / OUTD;
    const int Ni = in_sizes[1] / OUTD;
    const int E  = in_sizes[2] / 2;

    // ---- workspace carve-out (256B aligned) ----
    char* ws = (char*)d_ws;
    size_t off = 0;
    auto alloc = [&](size_t bytes) -> char* {
        char* p = ws + off;
        off = (off + bytes + 255) & ~(size_t)255;
        return p;
    };
    half_t* xu_h  = (half_t*)alloc((size_t)Nu * OUTD * 2);
    half_t* xi_h  = (half_t*)alloc((size_t)Ni * OUTD * 2);
    half_t* ku_h  = (half_t*)alloc((size_t)Nu * OUTD * 2);
    half_t* qu_h  = (half_t*)alloc((size_t)Nu * OUTD * 2);
    half_t* vu_h  = (half_t*)alloc((size_t)Nu * OUTD * 2);
    half_t* ki_h  = (half_t*)alloc((size_t)Ni * OUTD * 2);
    half_t* qi_h  = (half_t*)alloc((size_t)Ni * OUTD * 2);
    half_t* vi_h  = (half_t*)alloc((size_t)Ni * OUTD * 2);
    float*  au    = (float*)alloc((size_t)Nu * 4 * 4);
    float*  ai    = (float*)alloc((size_t)Ni * 4 * 4);
    size_t  zero0 = off;
    float*  msg_u = (float*)alloc((size_t)Nu * OUTD * 4);
    float*  msg_i = (float*)alloc((size_t)Ni * OUTD * 4);
    float*  wsu   = (float*)alloc((size_t)Nu * 4);
    float*  wsi   = (float*)alloc((size_t)Ni * 4);
    size_t  zero1 = off;
    half_t* agu_h = (half_t*)alloc((size_t)Nu * OUTD * 2);
    half_t* agi_h = (half_t*)alloc((size_t)Ni * OUTD * 2);
    half_t* tu_h  = (half_t*)alloc((size_t)Nu * OUTD * 2);
    half_t* ti_h  = (half_t*)alloc((size_t)Ni * OUTD * 2);
    half_t* fWk_u = (half_t*)alloc(128 * 128 * 2);
    half_t* fWq_u = (half_t*)alloc(128 * 128 * 2);
    half_t* fWv_u = (half_t*)alloc(128 * 128 * 2);
    half_t* fWk_i = (half_t*)alloc(128 * 128 * 2);
    half_t* fWq_i = (half_t*)alloc(128 * 128 * 2);
    half_t* fWv_i = (half_t*)alloc(128 * 128 * 2);
    half_t* fWm_u = (half_t*)alloc(128 * 128 * 2);
    half_t* fWm_i = (half_t*)alloc(128 * 128 * 2);
    half_t* fWg_u = (half_t*)alloc(128 * 128 * 2);
    half_t* fWg_i = (half_t*)alloc(128 * 128 * 2);
    (void)ws_size; (void)n_in; (void)out_size;

    float* out_user = (float*)d_out;
    float* out_item = (float*)d_out + (size_t)Nu * OUTD;

    // zero scatter targets (deterministic each launch; graph-capturable)
    (void)hipMemsetAsync(ws + zero0, 0, zero1 - zero0, stream);

    // weight shuffles + activation converts
    prep_wfrag<<<64, 256, 0, stream>>>(Wk_u, fWk_u);
    prep_wfrag<<<64, 256, 0, stream>>>(Wq_u, fWq_u);
    prep_wfrag<<<64, 256, 0, stream>>>(Wv_u, fWv_u);
    prep_wfrag<<<64, 256, 0, stream>>>(Wk_i, fWk_i);
    prep_wfrag<<<64, 256, 0, stream>>>(Wq_i, fWq_i);
    prep_wfrag<<<64, 256, 0, stream>>>(Wv_i, fWv_i);
    prep_wfrag<<<64, 256, 0, stream>>>(Wm_u, fWm_u);
    prep_wfrag<<<64, 256, 0, stream>>>(Wm_i, fWm_i);
    prep_wfrag<<<64, 256, 0, stream>>>(Wg_u, fWg_u);
    prep_wfrag<<<64, 256, 0, stream>>>(Wg_i, fWg_i);
    cvt_f32_f16<<<(Nu * OUTD + 255) / 256, 256, 0, stream>>>(x_user, xu_h, Nu * OUTD);
    cvt_f32_f16<<<(Ni * OUTD + 255) / 256, 256, 0, stream>>>(x_item, xi_h, Ni * OUTD);

    // K/Q/V projections (TDM-staged WMMA)
    const int gbU = (Nu + 63) / 64, gbI = (Ni + 63) / 64;
    gemm128_wmma_h<<<gbU, 256, 0, stream>>>(xu_h, fWk_u, nullptr, ku_h, Nu);
    gemm128_wmma_h<<<gbU, 256, 0, stream>>>(xu_h, fWq_u, nullptr, qu_h, Nu);
    gemm128_wmma_h<<<gbU, 256, 0, stream>>>(xu_h, fWv_u, nullptr, vu_h, Nu);
    gemm128_wmma_h<<<gbI, 256, 0, stream>>>(xi_h, fWk_i, nullptr, ki_h, Ni);
    gemm128_wmma_h<<<gbI, 256, 0, stream>>>(xi_h, fWq_i, nullptr, qi_h, Ni);
    gemm128_wmma_h<<<gbI, 256, 0, stream>>>(xi_h, fWv_i, nullptr, vi_h, Ni);

    // node-level attention bias (ek @ Wa hoisted out of the edge loop)
    attn_alpha<<<(Nu * 4 + 255) / 256, 256, 0, stream>>>(ku_h, Wa_ui, au, Nu);
    attn_alpha<<<(Ni * 4 + 255) / 256, 256, 0, stream>>>(ki_h, Wa_iu, ai, Ni);

    // edge message passing (one wave per edge, 8 edges/block)
    const int eb = (E + 7) / 8;
    edge_attn<<<eb, 256, 0, stream>>>(ku_h, vu_h, qi_h, au, ei_ui, E, msg_i, wsi); // dst=item
    edge_attn<<<eb, 256, 0, stream>>>(ki_h, vi_h, qu_h, ai, ei_iu, E, msg_u, wsu); // dst=user

    // normalize + convert
    agg_norm<<<(Nu * OUTD + 255) / 256, 256, 0, stream>>>(msg_u, wsu, agu_h, Nu);
    agg_norm<<<(Ni * OUTD + 255) / 256, 256, 0, stream>>>(msg_i, wsi, agi_h, Ni);

    // agg @ Wm + bm  (TDM-staged WMMA, f16 out)
    gemm128_wmma_h<<<gbU, 256, 0, stream>>>(agu_h, fWm_u, bm_u, tu_h, Nu);
    gemm128_wmma_h<<<gbI, 256, 0, stream>>>(agi_h, fWm_i, bm_i, ti_h, Ni);

    // (t @ Wg + bg) + residual, LayerNorm fused (TDM-staged WMMA)
    gemm128_wmma_ln<<<gbU, 256, 0, stream>>>(tu_h, fWg_u, bg_u, x_user, lng_u, lnb_u, out_user, Nu);
    gemm128_wmma_ln<<<gbI, 256, 0, stream>>>(ti_h, fWg_i, bg_i, x_item, lng_i, lnb_i, out_item, Ni);
}
